// Attention_63677185130777
// MI455X (gfx1250) — matrix operand order repaired and checked by reading, not compiled, not run on hardware
//
#include <hip/hip_runtime.h>

// ---------------- types ----------------
typedef __bf16 bf16;
typedef __attribute__((ext_vector_type(16))) __bf16 v16bf;
typedef __attribute__((ext_vector_type(8)))  __bf16 v8bf;
typedef __attribute__((ext_vector_type(8)))  float  v8f;
typedef __attribute__((ext_vector_type(4)))  int    v4i;

#define SEQ_   2048
#define DM_    2048
#define NH_    16
#define EH_    128        // d_head
#define IGNORE_ (-100000.0f)

// ---------------- CDNA5 async global->LDS copy (ASYNCcnt path) ----------------
#if defined(__has_builtin)
#if __has_builtin(__builtin_amdgcn_global_load_async_to_lds_b128)
#define HAS_ASYNC_LDS 1
#endif
#endif

#ifdef HAS_ASYNC_LDS
typedef __attribute__((address_space(1))) v4i gv4i;   // global int4
typedef __attribute__((address_space(3))) v4i lv4i;   // LDS int4
__device__ __forceinline__ void copy16_to_lds(const bf16* g, bf16* l) {
  // generic LDS pointer low 32 bits == LDS offset on AMDGPU
  __builtin_amdgcn_global_load_async_to_lds_b128(
      (gv4i*)(uintptr_t)g,
      (lv4i*)(uint32_t)(uintptr_t)l,
      /*offset=*/0, /*cpol=*/0);
}
__device__ __forceinline__ void wait_async_copies() {
#if __has_builtin(__builtin_amdgcn_s_wait_asynccnt)
  __builtin_amdgcn_s_wait_asynccnt(0);
#else
  asm volatile("s_wait_asynccnt 0" ::: "memory");
#endif
}
#else
__device__ __forceinline__ void copy16_to_lds(const bf16* g, bf16* l) {
  *(uint4*)l = *(const uint4*)g;   // global_load_b128 + ds_store_b128 fallback
}
__device__ __forceinline__ void wait_async_copies() {}
#endif

// ---------------- WMMA helpers (CDNA5 wave32) ----------------
__device__ __forceinline__ v8f vzero8() {
  v8f z;
#pragma unroll
  for (int i = 0; i < 8; ++i) z[i] = 0.0f;
  return z;
}

__device__ __forceinline__ v8f wmma_bf16(v16bf a, v16bf b, v8f c) {
  // D = A(16x32 bf16) * B(32x16 bf16) + C(16x16 f32)
  return __builtin_amdgcn_wmma_f32_16x16x32_bf16(
      /*neg_a=*/false, a, /*neg_b=*/false, b,
      /*c_mod=*/(short)0, c, /*reuse_a=*/false, /*reuse_b=*/false);
}

// A fragment: 16x32 (MxK), row-major source A[row][k], lda in elements.
// ISA layout: lanes 0-15 (M=lane): K=k0..k0+7 then k0+16..k0+23
//             lanes 16-31 (M=lane-16): K=k0+8..k0+15 then k0+24..k0+31
__device__ __forceinline__ v16bf load_a_frag(const bf16* A, int lda, int m0, int k0) {
  const int lane = threadIdx.x & 31;
  const int half = lane >> 4;
  const int row  = m0 + (lane & 15);
  const bf16* p = A + (size_t)row * lda + k0 + half * 8;
  v8bf lo = *(const v8bf*)(p);
  v8bf hi = *(const v8bf*)(p + 16);
  v16bf a;
#pragma unroll
  for (int i = 0; i < 8; ++i) { a[i] = lo[i]; a[i + 8] = hi[i]; }
  return a;
}

// B fragment: 32x16 (KxN) loaded from a TRANSPOSED store Bt[N][K] (ldb elems per row).
// ISA layout: lanes 0-15 (N=lane): K=k0..k0+7 then k0+16..k0+23 ; lanes 16-31 (N=lane-16): K=k0+8..k0+15 then k0+24..k0+31
__device__ __forceinline__ v16bf load_b_frag(const bf16* Bt, int ldb, int n0, int k0) {
  const int lane = threadIdx.x & 31;
  const int half = lane >> 4;
  const int col  = n0 + (lane & 15);
  const bf16* p = Bt + (size_t)col * ldb + k0 + half * 8;
  v8bf lo = *(const v8bf*)(p);
  v8bf hi = *(const v8bf*)(p + 16);
  v16bf b;
#pragma unroll
  for (int i = 0; i < 8; ++i) { b[i] = lo[i]; b[i + 8] = hi[i]; }
  return b;
}

// ---------------- 1) LayerNorm (no affine) -> bf16 ----------------
__global__ void ln_kernel(const float* __restrict__ x, bf16* __restrict__ xn) {
  __shared__ float s_sum[256];
  __shared__ float s_sq[256];
  const int row = blockIdx.x;
  const float* xr = x + (size_t)row * DM_;
  float v[8], sum = 0.f, sq = 0.f;
#pragma unroll
  for (int i = 0; i < 8; ++i) {
    v[i] = xr[threadIdx.x + i * 256];
    sum += v[i];
    sq  += v[i] * v[i];
  }
  s_sum[threadIdx.x] = sum;
  s_sq[threadIdx.x]  = sq;
  __syncthreads();
  for (int off = 128; off > 0; off >>= 1) {
    if ((int)threadIdx.x < off) {
      s_sum[threadIdx.x] += s_sum[threadIdx.x + off];
      s_sq[threadIdx.x]  += s_sq[threadIdx.x + off];
    }
    __syncthreads();
  }
  const float mean = s_sum[0] * (1.0f / DM_);
  const float var  = s_sq[0] * (1.0f / DM_) - mean * mean;
  const float rstd = rsqrtf(var + 1e-5f);
  bf16* o = xn + (size_t)row * DM_;
#pragma unroll
  for (int i = 0; i < 8; ++i)
    o[threadIdx.x + i * 256] = (bf16)((v[i] - mean) * rstd);
}

// ---------------- 2) weight convert/transposes to bf16 ----------------
// z=0..2: W_{Q,K,V}[h][d][e] -> wt[h][e][d] ; z=3: W_O[h][e][d] -> wot[d][h*128+e]
__global__ void convert_w_kernel(const float* __restrict__ wq, const float* __restrict__ wk,
                                 const float* __restrict__ wv, const float* __restrict__ wo,
                                 bf16* __restrict__ wtq, bf16* __restrict__ wtk,
                                 bf16* __restrict__ wtv, bf16* __restrict__ wot) {
  const int idx = blockIdx.x * 256 + threadIdx.x;   // 0 .. 4M-1
  const int z = blockIdx.y;
  if (z < 3) {
    const float* w = (z == 0) ? wq : (z == 1) ? wk : wv;
    bf16*       o = (z == 0) ? wtq : (z == 1) ? wtk : wtv;
    const int e = idx & 127;
    const int d = (idx >> 7) & 2047;
    const int h = idx >> 18;
    o[(((size_t)h * EH_ + e) << 11) + d] = (bf16)w[idx];
  } else {
    const int d = idx & 2047;
    const int e = (idx >> 11) & 127;
    const int h = idx >> 18;
    wot[((size_t)d << 11) + h * EH_ + e] = (bf16)wo[idx];
  }
}

// ---------------- 3) QKV projection GEMM + bias + fused rotary ----------------
// grid: (SEQ/64, NH, 3) ; block: 128 (4 waves). Wave computes 16 rows x 128 cols.
__global__ void qkv_kernel(const bf16* __restrict__ xn,
                           const bf16* __restrict__ wtq, const bf16* __restrict__ wtk,
                           const bf16* __restrict__ wtv,
                           const float* __restrict__ bq, const float* __restrict__ bk,
                           const float* __restrict__ bv,
                           bf16* __restrict__ qb, bf16* __restrict__ kb,
                           bf16* __restrict__ vt) {
  const int w    = threadIdx.x >> 5;
  const int lane = threadIdx.x & 31;
  const int half = lane >> 4;
  const int ln   = lane & 15;
  const int m0   = (blockIdx.x * 4 + w) * 16;
  const int h    = blockIdx.y;
  const int which = blockIdx.z;

  const bf16* Bt   = ((which == 0) ? wtq : (which == 1) ? wtk : wtv) + (size_t)h * EH_ * DM_;
  const float* bias = ((which == 0) ? bq : (which == 1) ? bk : bv) + h * EH_;

  v8f acc[8];
#pragma unroll
  for (int nt = 0; nt < 8; ++nt) acc[nt] = vzero8();

  for (int kc = 0; kc < DM_ / 32; ++kc) {
    const v16bf a = load_a_frag(xn, DM_, m0, kc * 32);
#pragma unroll
    for (int nt = 0; nt < 8; ++nt)
      acc[nt] = wmma_bf16(a, load_b_frag(Bt, DM_, nt * 16, kc * 32), acc[nt]);
  }

#pragma unroll
  for (int nt = 0; nt < 8; ++nt) {
    const int e = nt * 16 + ln;
    const float bval = bias[e];
    if (which < 2) {
      // rotary: angle = pos / 10000^{(e>>1)/64}; pairs (even,odd) live in lanes n, n^1
      const int   j    = e >> 1;
      const float invf = __powf(10000.0f, -(float)j * (1.0f / 64.0f));
      bf16* dst = (which == 0) ? qb : kb;
#pragma unroll
      for (int r = 0; r < 8; ++r) {
        const int   p       = m0 + r + half * 8;
        const float val     = acc[nt][r] + bval;
        const float partner = __shfl_xor(val, 1, 32);
        const float ang = (float)p * invf;
        const float sn = __sinf(ang), cs = __cosf(ang);
        const float rot = (e & 1) ? (val * cs + partner * sn)
                                  : (val * cs - partner * sn);
        dst[(((size_t)h * SEQ_ + p) << 7) + e] = (bf16)rot;
      }
    } else {
      // V stored transposed: vt[h][e][p]  (so P*V B-fragments are contiguous in keys)
#pragma unroll
      for (int r = 0; r < 8; ++r) {
        const int p = m0 + r + half * 8;
        vt[(((size_t)h * EH_ + e) << 11) + p] = (bf16)(acc[nt][r] + bval);
      }
    }
  }
}

// ---------------- 4) flash attention with async K/V LDS staging ----------------
// grid: (SEQ/64, NH) ; block 128 (4 waves). Block shares K/V tiles via double-buffered
// async global->LDS copies; each wave owns 16 query rows, streams 32 keys/step.
__global__ void attn_kernel(const bf16* __restrict__ qb, const bf16* __restrict__ kb,
                            const bf16* __restrict__ vt, bf16* __restrict__ zb) {
  __shared__ __align__(16) bf16 kbuf[2][32][EH_];   // [buf][key][e]   8KB x2
  __shared__ __align__(16) bf16 vbuf[2][EH_][32];   // [buf][e][key]   8KB x2
  __shared__ __align__(16) bf16 ptile[4][16][32];   // per-wave P bounce 4KB

  const int w    = threadIdx.x >> 5;
  const int lane = threadIdx.x & 31;
  const int half = lane >> 4;
  const int ln   = lane & 15;
  const int h    = blockIdx.y;
  const int q0b  = blockIdx.x * 64;
  const int q0   = q0b + w * 16;

  const bf16* Q  = qb + (size_t)h * SEQ_ * EH_;   // [p][e]
  const bf16* K  = kb + (size_t)h * SEQ_ * EH_;   // [p][e]
  const bf16* Vt = vt + (size_t)h * EH_ * SEQ_;   // [e][p]

  v16bf qf[4];
#pragma unroll
  for (int c = 0; c < 4; ++c) qf[c] = load_a_frag(Q, EH_, q0, c * 32);

  v8f   zacc[8];
  float m[8], l[8];
#pragma unroll
  for (int c = 0; c < 8; ++c) zacc[c] = vzero8();
#pragma unroll
  for (int r = 0; r < 8; ++r) { m[r] = -3.0e38f; l[r] = 0.0f; }

  const float inv_scale = 0.0883883476483184f;    // 1/sqrt(128)
  const int nsteps = (q0b + 63) / 32 + 1;         // block-uniform causal cap

  // --- tile staging: 128 threads move K (32x128) + V (128x32) = 16KB/step ---
  const int t = threadIdx.x;
  auto stage = [&](int st, int buf) {
    const int kt = st * 32;
#pragma unroll
    for (int i = 0; i < 4; ++i) {           // K: 512 16B-chunks
      const int c = t + i * 128;
      const int row = c >> 4, col = (c & 15) * 8;
      copy16_to_lds(K + (size_t)(kt + row) * EH_ + col, &kbuf[buf][row][col]);
    }
#pragma unroll
    for (int i = 0; i < 4; ++i) {           // V: 512 16B-chunks
      const int c = t + i * 128;
      const int row = c >> 2, col = (c & 3) * 8;
      copy16_to_lds(Vt + (size_t)row * SEQ_ + kt + col, &vbuf[buf][row][col]);
    }
  };

  stage(0, 0);
  wait_async_copies();
  __syncthreads();

  for (int st = 0; st < nsteps; ++st) {
    const int cur = st & 1;
    if (st + 1 < nsteps) stage(st + 1, cur ^ 1);   // prefetch next tile while computing

    const int kt0 = st * 32;
    v8f s0 = vzero8(), s1 = vzero8();
#pragma unroll
    for (int c = 0; c < 4; ++c) {
      s0 = wmma_bf16(qf[c], load_b_frag(&kbuf[cur][0][0], EH_, 0,  c * 32), s0);
      s1 = wmma_bf16(qf[c], load_b_frag(&kbuf[cur][0][0], EH_, 16, c * 32), s1);
    }
    const int col0 = kt0 + ln, col1 = col0 + 16;
    float facs[8];
#pragma unroll
    for (int r = 0; r < 8; ++r) {
      const int row = q0 + r + half * 8;
      float a0 = s0[r] * inv_scale;
      float a1 = s1[r] * inv_scale;
      if (col0 > row) a0 = IGNORE_;
      if (col1 > row) a1 = IGNORE_;
      // row max across the 16 lanes of this half (C-layout row striping)
      float mt = fmaxf(a0, a1);
#pragma unroll
      for (int msk = 1; msk < 16; msk <<= 1) mt = fmaxf(mt, __shfl_xor(mt, msk, 32));
      const float mn = fmaxf(m[r], mt);
      const float p0 = __expf(a0 - mn);
      const float p1 = __expf(a1 - mn);
      float rs = p0 + p1;
#pragma unroll
      for (int msk = 1; msk < 16; msk <<= 1) rs += __shfl_xor(rs, msk, 32);
      const float fac = __expf(m[r] - mn);
      m[r] = mn;
      l[r] = l[r] * fac + rs;
      facs[r] = fac;
      s0[r] = p0;
      s1[r] = p1;
    }
#pragma unroll
    for (int c = 0; c < 8; ++c)
#pragma unroll
      for (int r = 0; r < 8; ++r) zacc[c][r] *= facs[r];

    // P (C-layout) -> LDS -> A-fragment. Per-wave LDS; DS ops are in-order per wave.
#pragma unroll
    for (int r = 0; r < 8; ++r) {
      ptile[w][r + half * 8][ln]      = (bf16)s0[r];
      ptile[w][r + half * 8][ln + 16] = (bf16)s1[r];
    }
    asm volatile("" ::: "memory");
    const v16bf pa = load_a_frag(&ptile[w][0][0], 32, 0, 0);
#pragma unroll
    for (int c = 0; c < 8; ++c)
      zacc[c] = wmma_bf16(pa, load_b_frag(&vbuf[cur][0][0], 32, c * 16, 0), zacc[c]);

    wait_async_copies();   // next tile landed
    __syncthreads();       // all waves done reading cur; next tile visible block-wide
  }

  // normalize and write z[p][h*128+e] (bf16, feeds output GEMM)
#pragma unroll
  for (int c = 0; c < 8; ++c)
#pragma unroll
    for (int r = 0; r < 8; ++r) {
      const int p = q0 + r + half * 8;
      zb[((size_t)p << 11) + h * EH_ + c * 16 + ln] = (bf16)(zacc[c][r] / l[r]);
    }
}

// ---------------- 5) output projection: out = z @ W_O + b_O ----------------
// grid: (SEQ/64, DM/64) ; block 128. Wave: 16 rows x 64 cols.
__global__ void out_kernel(const bf16* __restrict__ zb, const bf16* __restrict__ wot,
                           const float* __restrict__ bo, float* __restrict__ out) {
  const int w    = threadIdx.x >> 5;
  const int lane = threadIdx.x & 31;
  const int half = lane >> 4;
  const int ln   = lane & 15;
  const int m0   = (blockIdx.x * 4 + w) * 16;
  const int n0   = blockIdx.y * 64;

  v8f acc[4];
#pragma unroll
  for (int nt = 0; nt < 4; ++nt) acc[nt] = vzero8();

  for (int kc = 0; kc < DM_ / 32; ++kc) {
    const v16bf a = load_a_frag(zb, DM_, m0, kc * 32);
#pragma unroll
    for (int nt = 0; nt < 4; ++nt)
      acc[nt] = wmma_bf16(a, load_b_frag(wot, DM_, n0 + nt * 16, kc * 32), acc[nt]);
  }
#pragma unroll
  for (int nt = 0; nt < 4; ++nt) {
    const int d = n0 + nt * 16 + ln;
    const float bb = bo[d];
#pragma unroll
    for (int r = 0; r < 8; ++r) {
      const int p = m0 + r + half * 8;
      out[((size_t)p << 11) + d] = acc[nt][r] + bb;
    }
  }
}

// ---------------- host launcher ----------------
extern "C" void kernel_launch(void* const* d_in, const int* in_sizes, int n_in,
                              void* d_out, int out_size, void* d_ws, size_t ws_size,
                              hipStream_t stream) {
  const float* resid = (const float*)d_in[0];
  const float* W_Q   = (const float*)d_in[1];
  const float* W_K   = (const float*)d_in[2];
  const float* W_V   = (const float*)d_in[3];
  const float* W_O   = (const float*)d_in[4];
  const float* b_Q   = (const float*)d_in[5];
  const float* b_K   = (const float*)d_in[6];
  const float* b_V   = (const float*)d_in[7];
  const float* b_O   = (const float*)d_in[8];
  float* out = (float*)d_out;

  char* ws = (char*)d_ws;
  const size_t MB = 1024ull * 1024ull;
  bf16* xn  = (bf16*)(ws + 0 * MB);    // [2048][2048]
  bf16* wtq = (bf16*)(ws + 8 * MB);    // [h][e][d]
  bf16* wtk = (bf16*)(ws + 16 * MB);
  bf16* wtv = (bf16*)(ws + 24 * MB);
  bf16* wot = (bf16*)(ws + 32 * MB);   // [d][h*e]
  bf16* qb  = (bf16*)(ws + 40 * MB);   // [h][p][e] (rotary applied)
  bf16* kb  = (bf16*)(ws + 48 * MB);   // [h][p][e] (rotary applied)
  bf16* vt  = (bf16*)(ws + 56 * MB);   // [h][e][p]
  bf16* zb  = (bf16*)(ws + 64 * MB);   // [p][h*e]

  ln_kernel<<<SEQ_, 256, 0, stream>>>(resid, xn);
  convert_w_kernel<<<dim3((NH_ * DM_ * EH_) / 256, 4), 256, 0, stream>>>(
      W_Q, W_K, W_V, W_O, wtq, wtk, wtv, wot);
  qkv_kernel<<<dim3(SEQ_ / 64, NH_, 3), 128, 0, stream>>>(
      xn, wtq, wtk, wtv, b_Q, b_K, b_V, qb, kb, vt);
  attn_kernel<<<dim3(SEQ_ / 64, NH_), 128, 0, stream>>>(qb, kb, vt, zb);
  out_kernel<<<dim3(SEQ_ / 64, DM_ / 64), 128, 0, stream>>>(zb, wot, b_O, out);
}
